// CustomAttention_10900626997652
// MI455X (gfx1250) — matrix operand order repaired and checked by reading, not compiled, not run on hardware
//
#include <hip/hip_runtime.h>

// ---------------------------------------------------------------------------
// Flash-attention forward for gfx1250 (MI455X), wave32 + v_wmma_f32_16x16x32_f16
// B=2, S=2048, H=16, D=64, fp32 I/O, f16 matrix operands, f32 accumulation.
// Register double-buffered K/V staging: next block's global loads are issued
// before the current block's WMMA pipeline, hiding HBM/L2 latency.
// ---------------------------------------------------------------------------

typedef __attribute__((ext_vector_type(16))) _Float16 v16h;
typedef __attribute__((ext_vector_type(8)))  _Float16 v8h;
typedef __attribute__((ext_vector_type(4)))  _Float16 v4h;
typedef __attribute__((ext_vector_type(8)))  float    v8f;

#define BATCH    2
#define SEQ      2048
#define N_HEAD   16
#define HEAD_DIM 64
#define EMB      1024
#define SCALE    0.125f      // 1/sqrt(64)

#define BM     128           // query rows per workgroup (8 waves x 16)
#define BN     64            // key rows per inner iteration
#define NWAVE  8
#define LDH    72            // LDS row stride in halves (144B: 16B-aligned rows, bank-skewed)

union AFrag { v16h v; v8h h[2]; };

__device__ __forceinline__ v8h cvt8(const float4 a, const float4 b, const float s) {
  v8h r;
  r[0] = (_Float16)(a.x * s); r[1] = (_Float16)(a.y * s);
  r[2] = (_Float16)(a.z * s); r[3] = (_Float16)(a.w * s);
  r[4] = (_Float16)(b.x * s); r[5] = (_Float16)(b.y * s);
  r[6] = (_Float16)(b.z * s); r[7] = (_Float16)(b.w * s);
  return r;
}

__global__ __launch_bounds__(256)
void flash_attn_fwd_gfx1250(const float* __restrict__ Q,
                            const float* __restrict__ K,
                            const float* __restrict__ V,
                            const int*   __restrict__ AM,
                            float* __restrict__ O)
{
  // LDS: K block row-major [key][d], V block transposed [d][key], P per-wave [row][key]
  __shared__ _Float16 Ks[BN * LDH];              //  9216 B
  __shared__ _Float16 Vt[HEAD_DIM * LDH];        //  9216 B
  __shared__ _Float16 Ps[NWAVE * 16 * LDH];      // 18432 B

  const int tid  = threadIdx.x;
  const int wid  = tid >> 5;
  const int lane = tid & 31;
  const int l16  = lane & 15;
  const int hi   = (lane >> 4) & 1;              // lane half: 0 -> 0..15, 1 -> 16..31

  const int qtile = blockIdx.x;                  // 0..15
  const int bh    = blockIdx.y;                  // 0..31
  const int b     = bh / N_HEAD;
  const int h     = bh % N_HEAD;

  const int q0 = qtile * BM;
  const int qrow = q0 + wid * 16 + l16;

  const size_t headoff = (size_t)h * HEAD_DIM;
  const float* qrow_ptr = Q + ((size_t)b * SEQ + qrow) * EMB + headoff;
  const float* kbase    = K + ((size_t)b * SEQ) * EMB + headoff;
  const float* vbase    = V + ((size_t)b * SEQ) * EMB + headoff;
  const int*   mbase    = AM + b * SEQ;

  // Staging geometry: 1024 float4 slots per 64x64 block, 4 per thread.
  const int sr = tid >> 4;                       // base key row 0..15 (row = sr + i*16)
  const int sc = (tid & 15) << 2;                // d column 0..60

  // ---- Q tile -> A-fragment registers (16x32 f16 layout), pre-scaled ------
  // Lanes 0-15 hold row M=l16 with K {0..7,16..23}; lanes 16-31 K {8..15,24..31}.
  AFrag aq[2];
  {
    const int sh = hi ? 8 : 0;
    #pragma unroll
    for (int f = 0; f < 2; ++f) {
      const float4 p0 = *reinterpret_cast<const float4*>(qrow_ptr + f * 32 + sh);
      const float4 p1 = *reinterpret_cast<const float4*>(qrow_ptr + f * 32 + sh + 4);
      const float4 p2 = *reinterpret_cast<const float4*>(qrow_ptr + f * 32 + 16 + sh);
      const float4 p3 = *reinterpret_cast<const float4*>(qrow_ptr + f * 32 + 16 + sh + 4);
      aq[f].h[0] = cvt8(p0, p1, SCALE);          // fold softmax scale into Q
      aq[f].h[1] = cvt8(p2, p3, SCALE);
    }
  }

  // ---- flash accumulators -------------------------------------------------
  const v8f vzero = {};
  v8f   o0[4];                                   // 16x64 context tile (4 C-frags)
  float m_i[8], l_i[8];                          // per-row running max / sum
  #pragma unroll
  for (int j = 0; j < 4; ++j) o0[j] = vzero;
  #pragma unroll
  for (int r = 0; r < 8; ++r) { m_i[r] = -1.0e30f; l_i[r] = 0.0f; }

  // Double-buffer registers for the next K/V block.
  float4 nk[4], nv[4];

  // ---- prologue: fetch block 0 into regs, stage into LDS ------------------
  #pragma unroll
  for (int i = 0; i < 4; ++i) {
    const int r = sr + i * 16;
    nk[i] = *reinterpret_cast<const float4*>(kbase + (size_t)r * EMB + sc);
    nv[i] = *reinterpret_cast<const float4*>(vbase + (size_t)r * EMB + sc);
  }
  #pragma unroll
  for (int i = 0; i < 4; ++i) {
    const int r = sr + i * 16;
    v4h hk;
    hk[0] = (_Float16)nk[i].x; hk[1] = (_Float16)nk[i].y;
    hk[2] = (_Float16)nk[i].z; hk[3] = (_Float16)nk[i].w;
    *reinterpret_cast<v4h*>(&Ks[r * LDH + sc]) = hk;
    Vt[(sc + 0) * LDH + r] = (_Float16)nv[i].x;
    Vt[(sc + 1) * LDH + r] = (_Float16)nv[i].y;
    Vt[(sc + 2) * LDH + r] = (_Float16)nv[i].z;
    Vt[(sc + 3) * LDH + r] = (_Float16)nv[i].w;
  }
  __syncthreads();

  for (int n0 = 0; n0 < SEQ; n0 += BN) {
    const bool has_next = (n0 + BN) < SEQ;

    // ---- issue next block's global loads NOW (latency hidden by compute) ---
    if (has_next) {
      #pragma unroll
      for (int i = 0; i < 4; ++i) {
        const int r = n0 + BN + sr + i * 16;
        nk[i] = *reinterpret_cast<const float4*>(kbase + (size_t)r * EMB + sc);
        nv[i] = *reinterpret_cast<const float4*>(vbase + (size_t)r * EMB + sc);
      }
      // warm L2 two blocks ahead (global_prefetch_b8)
      if (n0 + 2 * BN < SEQ) {
        __builtin_prefetch(kbase + (size_t)(n0 + 2 * BN + sr) * EMB + sc, 0, 1);
        __builtin_prefetch(vbase + (size_t)(n0 + 2 * BN + sr) * EMB + sc, 0, 1);
      }
    }

    // ---- S = (Q*scale) @ K^T : 4 N-tiles x 2 K-chunks = 8 WMMA -------------
    v8f s[4];
    #pragma unroll
    for (int j = 0; j < 4; ++j) {
      v8f acc = vzero;
      #pragma unroll
      for (int f = 0; f < 2; ++f) {
        AFrag bk;                                // B: column = key row (contig in LDS)
        const _Float16* src = &Ks[(j * 16 + l16) * LDH + f * 32 + hi * 8];
        bk.h[0] = *reinterpret_cast<const v8h*>(src);
        bk.h[1] = *reinterpret_cast<const v8h*>(src + 16);
        acc = __builtin_amdgcn_wmma_f32_16x16x32_f16(false, aq[f].v, false, bk.v,
                                                     (short)0, acc, false, false);
      }
      s[j] = acc;
    }

    // ---- mask bias (branchless: keep EXEC all-ones for WMMA) ---------------
    float bias[4];
    #pragma unroll
    for (int j = 0; j < 4; ++j)
      bias[j] = (mbase[n0 + j * 16 + l16] == 0) ? -1.0e30f : 0.0f;

    // ---- online softmax; row data lives across one 16-lane half ------------
    #pragma unroll
    for (int r = 0; r < 8; ++r) {
      float v0 = s[0][r] + bias[0];
      float v1 = s[1][r] + bias[1];
      float v2 = s[2][r] + bias[2];
      float v3 = s[3][r] + bias[3];
      float rm = fmaxf(fmaxf(v0, v1), fmaxf(v2, v3));
      #pragma unroll
      for (int off = 8; off >= 1; off >>= 1)
        rm = fmaxf(rm, __shfl_xor(rm, off, 32));
      const float mnew = fmaxf(m_i[r], rm);
      const float corr = __expf(m_i[r] - mnew);
      const float p0 = __expf(v0 - mnew);
      const float p1 = __expf(v1 - mnew);
      const float p2 = __expf(v2 - mnew);
      const float p3 = __expf(v3 - mnew);
      s[0][r] = p0; s[1][r] = p1; s[2][r] = p2; s[3][r] = p3;
      float rs = (p0 + p1) + (p2 + p3);
      #pragma unroll
      for (int off = 8; off >= 1; off >>= 1)
        rs += __shfl_xor(rs, off, 32);
      l_i[r] = l_i[r] * corr + rs;
      m_i[r] = mnew;
      o0[0][r] *= corr; o0[1][r] *= corr; o0[2][r] *= corr; o0[3][r] *= corr;
    }

    // ---- P (C-layout) -> LDS -> A-layout for the PV GEMM -------------------
    #pragma unroll
    for (int r = 0; r < 8; ++r) {
      const int row = wid * 16 + hi * 8 + r;
      #pragma unroll
      for (int j = 0; j < 4; ++j)
        Ps[row * LDH + j * 16 + l16] = (_Float16)s[j][r];
    }
    __syncthreads();

    // ---- O += P @ V : 2 K-chunks x 4 D-tiles = 8 WMMA ----------------------
    #pragma unroll
    for (int f = 0; f < 2; ++f) {
      AFrag ap;
      const _Float16* prow = &Ps[(wid * 16 + l16) * LDH + f * 32];
      ap.h[0] = *reinterpret_cast<const v8h*>(prow + hi * 8);
      ap.h[1] = *reinterpret_cast<const v8h*>(prow + 16 + hi * 8);
      #pragma unroll
      for (int jd = 0; jd < 4; ++jd) {
        AFrag bv;                                // B: column = head-dim (contig in Vt)
        const _Float16* vcol = &Vt[(jd * 16 + l16) * LDH + f * 32 + hi * 8];
        bv.h[0] = *reinterpret_cast<const v8h*>(vcol);
        bv.h[1] = *reinterpret_cast<const v8h*>(vcol + 16);
        o0[jd] = __builtin_amdgcn_wmma_f32_16x16x32_f16(false, ap.v, false, bv.v,
                                                        (short)0, o0[jd], false, false);
      }
    }

    // ---- rotate double buffer: stage prefetched regs into LDS --------------
    if (has_next) {
      __syncthreads();                           // everyone done reading Ks/Vt
      #pragma unroll
      for (int i = 0; i < 4; ++i) {
        const int r = sr + i * 16;
        v4h hk;
        hk[0] = (_Float16)nk[i].x; hk[1] = (_Float16)nk[i].y;
        hk[2] = (_Float16)nk[i].z; hk[3] = (_Float16)nk[i].w;
        *reinterpret_cast<v4h*>(&Ks[r * LDH + sc]) = hk;
        Vt[(sc + 0) * LDH + r] = (_Float16)nv[i].x;
        Vt[(sc + 1) * LDH + r] = (_Float16)nv[i].y;
        Vt[(sc + 2) * LDH + r] = (_Float16)nv[i].z;
        Vt[(sc + 3) * LDH + r] = (_Float16)nv[i].w;
      }
      __syncthreads();                           // new block visible
    }
  }

  // ---- normalize and store ------------------------------------------------
  float* obase = O + ((size_t)b * SEQ + q0 + wid * 16) * EMB + headoff;
  #pragma unroll
  for (int r = 0; r < 8; ++r) {
    const int row = hi * 8 + r;
    const float inv = 1.0f / l_i[r];
    #pragma unroll
    for (int jd = 0; jd < 4; ++jd)
      obase[(size_t)row * EMB + jd * 16 + l16] = o0[jd][r] * inv;
  }
}

extern "C" void kernel_launch(void* const* d_in, const int* in_sizes, int n_in,
                              void* d_out, int out_size, void* d_ws, size_t ws_size,
                              hipStream_t stream) {
  (void)in_sizes; (void)n_in; (void)out_size; (void)d_ws; (void)ws_size;
  const float* Q  = (const float*)d_in[0];
  const float* K  = (const float*)d_in[1];
  const float* V  = (const float*)d_in[2];
  const int*   AM = (const int*)d_in[3];
  float* O = (float*)d_out;
  dim3 grid(SEQ / BM, BATCH * N_HEAD);   // 16 x 32 = 512 workgroups
  flash_attn_fwd_gfx1250<<<grid, 256, 0, stream>>>(Q, K, V, AM, O);
}
